// RecurrentSheafLayer_36507222016695
// MI455X (gfx1250) — compile-verified, matrix operand order checked by reading, not run on hardware
//
#include <hip/hip_runtime.h>
#include <hip/hip_bf16.h>
#include <math.h>

// Problem constants (reference: B=8, L=4096, D=1024)
static constexpr int Bb = 8;
static constexpr int Ll = 4096;
static constexpr int Dd = 1024;
static constexpr int Mrows = Bb * Ll; // 32768 flattened (b,l) rows

typedef __attribute__((ext_vector_type(16))) __bf16 v16bf;
typedef __attribute__((ext_vector_type(8)))  float  v8f;
typedef __attribute__((ext_vector_type(4)))  int    v4i;

#define DEV __device__ __forceinline__

// ---------------------------------------------------------------------------
// CDNA5 16-bit A/B fragment swizzle. For a 32-wide K chunk:
//   hardware lane = (row & 15) + 16*half(k),  half(k) = (k>>3)&1
//   element      = (k & 7) | ((k & 16) >> 1)
// Global swizzled layout: [row/16 tile][k/32 chunk][lane 32][elem 16]
// -> one 16-row x 1024-K tile = 16384 bf16 = 32 KB contiguous.
// ---------------------------------------------------------------------------
DEV int khalf(int k) { return (k >> 3) & 1; }
DEV int kswz(int k)  { return (k & 7) | ((k & 16) >> 1); }
DEV size_t aswz(int row, int k) {
  return (size_t)(row >> 4) * (32 * 512) + (size_t)(k >> 5) * 512 +
         (size_t)(((row & 15) + 16 * khalf(k)) * 16 + kswz(k));
}

DEV float sigmoidf(float v) { return 1.0f / (1.0f + __expf(-v)); }

DEV v8f wmma_bf16(const v16bf& a, const v16bf& b, const v8f& c) {
  return __builtin_amdgcn_wmma_f32_16x16x32_bf16(false, a, false, b,
                                                 (short)0, c, false, false);
}

// ---------------------------------------------------------------------------
// Async global->LDS copy (CDNA5 GLOBAL_LOAD_ASYNC_TO_LDS_B128, ASYNCcnt),
// with a portable fallback. Builtin signature (from hipcc diagnostics):
//   (v4i addrspace(1)*, v4i addrspace(3)*, imm int offset, imm int cpol)
// ---------------------------------------------------------------------------
#if defined(__has_builtin)
#if __has_builtin(__builtin_amdgcn_global_load_async_to_lds_b128) && \
    __has_builtin(__builtin_amdgcn_s_wait_asynccnt)
#define USE_ASYNC_LDS 1
#endif
#endif
#ifndef USE_ASYNC_LDS
#define USE_ASYNC_LDS 0
#endif

#if USE_ASYNC_LDS
typedef __attribute__((address_space(1))) v4i* gv4i_p;
typedef __attribute__((address_space(3))) v4i* lv4i_p;
#endif

DEV void copy16(__bf16* ldst, const __bf16* gsrc) {
#if USE_ASYNC_LDS
  __builtin_amdgcn_global_load_async_to_lds_b128((gv4i_p)gsrc, (lv4i_p)ldst, 0, 0);
#else
  *reinterpret_cast<int4*>(ldst) = *reinterpret_cast<const int4*>(gsrc);
#endif
}
DEV void async_join() {
#if USE_ASYNC_LDS
  __builtin_amdgcn_s_wait_asynccnt(0);
#endif
}

// ---------------------------------------------------------------------------
// f32 -> bf16 conversion with fragment swizzle (weights: 1024x1024)
// ---------------------------------------------------------------------------
__global__ void to_bf16_swz_kernel(const float* __restrict__ src,
                                   __bf16* __restrict__ dst) {
  for (int i = (blockIdx.x * blockDim.x + threadIdx.x) * 4; i < Dd * Dd;
       i += gridDim.x * blockDim.x * 4) {
    const int r = i >> 10, k = i & 1023; // k multiple of 4
    const float4 v = *reinterpret_cast<const float4*>(src + i);
    __bf16* d = dst + aswz(r, k); // 4 consecutive slots
    d[0] = (__bf16)v.x; d[1] = (__bf16)v.y;
    d[2] = (__bf16)v.z; d[3] = (__bf16)v.w;
  }
}

// Zero h ping-pong buffers and the global barrier counter (every launch).
__global__ void reset_kernel(float* __restrict__ hbuf, unsigned* __restrict__ cnt) {
  int i = blockIdx.x * blockDim.x + threadIdx.x;
  if (i < 2 * Bb * Dd) hbuf[i] = 0.0f;
  if (i == 0) *cnt = 0u;
}

// ---------------------------------------------------------------------------
// Tiled bf16 WMMA GEMM:  out[m,n] = epilogue( sum_k A[m,k]*W[n,k] + bias[n] )
// W is pre-swizzled (fragment layout). MODE 0: A = f32 x (convert+swizzle on
// the fly), epilogue = sigmoid -> bf16 z. MODE 1: A = pre-swizzled bf16 y_ln,
// staged with contiguous (async) copies, epilogue = identity -> f32.
// Block tile 32(M) x 64(N); 8 waves; K staged in chunks of 128.
// ---------------------------------------------------------------------------
template <int MODE>
__launch_bounds__(256, 2)
__global__ void gemm_kernel(const float* __restrict__ Af32,
                            const __bf16* __restrict__ Abf_swz,
                            const __bf16* __restrict__ Wswz,
                            const float* __restrict__ bias,
                            __bf16* __restrict__ out_bf,
                            float* __restrict__ out_f32) {
  constexpr int KC  = 128;      // K per staged block
  constexpr int KCH = KC / 32;  // 32-wide chunks per block
  __shared__ alignas(16) __bf16 as_s[2 * KCH * 512]; //  8 KB [mt][chunk][lane][16]
  __shared__ alignas(16) __bf16 bs_s[4 * KCH * 512]; // 16 KB [nt][chunk][lane][16]

  const int tid  = threadIdx.x;
  const int lane = tid & 31;
  const int wave = tid >> 5;
  const int mt   = wave & 1;   // 0..1  -> M sub-tile
  const int nt   = wave >> 1;  // 0..3  -> N sub-tile
  const int m0   = blockIdx.x * 32;
  const int n0   = blockIdx.y * 64;
  const int nl   = lane & 15;
  const int half = lane >> 4;

  // Staging coordinates for the MODE 0 (f32 x) path
  const int arow = tid >> 3;        // A row 0..31, 16 consecutive k each
  const int acol = (tid & 7) * 16;

  v8f acc0 = {}, acc1 = {};

  for (int k0 = 0; k0 < Dd; k0 += KC) {
    // ---- stage A tile 32xKC ----
    if constexpr (MODE == 0) {
      const int amt = arow >> 4, arl = arow & 15;
      const float* src = Af32 + (size_t)(m0 + arow) * Dd + k0 + acol;
      if (k0 + KC < Dd) __builtin_prefetch(src + KC, 0, 1);
#pragma unroll
      for (int g = 0; g < 4; ++g) {
        const int k = acol + g * 4;
        const float4 v = *reinterpret_cast<const float4*>(src + g * 4);
        __bf16* dst = &as_s[(size_t)amt * (KCH * 512) + (k >> 5) * 512 +
                            (arl + 16 * khalf(k)) * 16 + kswz(k)];
        dst[0] = (__bf16)v.x; dst[1] = (__bf16)v.y;
        dst[2] = (__bf16)v.z; dst[3] = (__bf16)v.w;
      }
    } else {
      // Pre-swizzled A: 2 contiguous runs of 2048 elems (one per m-tile)
      for (int idx = tid * 8; idx < 2 * KCH * 512; idx += 256 * 8) {
        const int mtr = idx >> 11, within = idx & 2047;
        copy16(&as_s[mtr * 2048 + within],
               Abf_swz + (size_t)((m0 >> 4) + mtr) * 16384 + (k0 >> 5) * 512 + within);
      }
    }
    // ---- stage B tile 64xKC: 4 contiguous runs of 2048 elems ----
    for (int idx = tid * 8; idx < 4 * KCH * 512; idx += 256 * 8) {
      const int ntr = idx >> 11, within = idx & 2047;
      copy16(&bs_s[ntr * 2048 + within],
             Wswz + (size_t)((n0 >> 4) + ntr) * 16384 + (k0 >> 5) * 512 + within);
    }
    async_join();
    __syncthreads();

#pragma unroll
    for (int ch = 0; ch < KCH; ++ch) {
      const v16bf a =
          *reinterpret_cast<const v16bf*>(&as_s[((mt * KCH + ch) * 32 + lane) * 16]);
      const v16bf b =
          *reinterpret_cast<const v16bf*>(&bs_s[((nt * KCH + ch) * 32 + lane) * 16]);
      if (ch & 1) acc1 = wmma_bf16(a, b, acc1);
      else        acc0 = wmma_bf16(a, b, acc0);
    }
    __syncthreads();
  }

  const v8f acc = acc0 + acc1;

  // Epilogue: C/D layout -> lane nl = N col, VGPR r = M row (+8 for lanes 16-31)
  const int n = n0 + nt * 16 + nl;
  const float bn = bias[n];
#pragma unroll
  for (int r = 0; r < 8; ++r) {
    const size_t m = (size_t)m0 + mt * 16 + r + 8 * half;
    const float v = acc[r] + bn;
    if constexpr (MODE == 0)
      out_bf[m * Dd + n] = (__bf16)sigmoidf(v);
    else
      out_f32[m * Dd + n] = v;
  }
}

// ---------------------------------------------------------------------------
// Persistent recurrent scan. 64 blocks x 128 threads; block nb owns output
// columns [nb*16, nb*16+16). Its pre-swizzled 32 KB Wr slice is one
// contiguous run, loaded (async) into LDS once and resident for all 4096
// steps. Per step: stage h (f32->bf16, swizzled), one wave runs 32 WMMAs
// (4 accumulator chains) with the epilogue's global loads hoisted above the
// matrix work; wave 1 prefetches x/z 8 steps ahead; then a device-wide
// monotonic atomic barrier.
// ---------------------------------------------------------------------------
__launch_bounds__(128, 1)
__global__ void scan_kernel(const float* __restrict__ x,
                            const float* __restrict__ decay,
                            const float* __restrict__ br,
                            const __bf16* __restrict__ Wr_swz,
                            const __bf16* __restrict__ zbuf,
                            float* __restrict__ hbuf,
                            unsigned* __restrict__ cnt,
                            float* __restrict__ ybuf) {
  __shared__ alignas(16) __bf16 wr_sw[32 * 512]; // 32 KB [chunk][lane][16]
  __shared__ alignas(16) __bf16 h_sw [32 * 512]; // 32 KB [chunk][lane][16]

  const int nb   = blockIdx.x;  // 64
  const int n0   = nb * 16;
  const int tid  = threadIdx.x; // 128
  const int lane = tid & 31;
  const int nl   = lane & 15;
  const int half = lane >> 4;
  const int ncol = n0 + nl;
  const unsigned nblk = gridDim.x;

  // Zero the swizzled h buffer once (covers the M=8..15 padding lanes).
  for (int i = tid; i < 32 * 512; i += 128) h_sw[i] = (__bf16)0.0f;

  // Resident Wr slice: tile nb of the pre-swizzled Wr = contiguous 16384 elems.
  {
    const __bf16* wsrc = Wr_swz + (size_t)nb * 16384;
    for (int idx = tid * 8; idx < 16384; idx += 128 * 8)
      copy16(&wr_sw[idx], wsrc + idx);
    async_join();
  }

  const float df  = sigmoidf(decay[ncol]);
  const float omd = 1.0f - df;
  const float brn = br[ncol];

  for (int t = 0; t < Ll; ++t) {
    const float* hcur = hbuf + (size_t)(t & 1) * (Bb * Dd);
    float*       hnxt = hbuf + (size_t)((t + 1) & 1) * (Bb * Dd);

    // Stage h (8x1024 f32 from L2) into the swizzled bf16 buffer.
    for (int i = tid * 4; i < Bb * Dd; i += 128 * 4) {
      const int m = i >> 10, k = i & 1023; // k multiple of 4
      const float* src = hcur + i;
      __bf16* dst = &h_sw[((k >> 5) * 32 + m + 16 * khalf(k)) * 16 + kswz(k)];
#pragma unroll
      for (int j = 0; j < 4; ++j) dst[j] = (__bf16)src[j];
    }
    __syncthreads();

    if (tid < 32) { // wave 0: pred[8,16] = h @ Wr_slice^T
      // Hoist the epilogue's global loads above the WMMA chain.
      float xv[8], zv[8], hold[8];
      if (half == 0) {
#pragma unroll
        for (int r = 0; r < 8; ++r) {
          const size_t row = (size_t)r * Ll + t;
          xv[r]   = x[row * Dd + ncol];
          zv[r]   = (float)zbuf[row * Dd + ncol];
          hold[r] = hcur[r * Dd + ncol];
        }
      }

      v8f accs[4] = {{}, {}, {}, {}};
#pragma unroll
      for (int ch = 0; ch < 32; ++ch) {
        const v16bf a = *reinterpret_cast<const v16bf*>(&h_sw[(ch * 32 + lane) * 16]);
        const v16bf b = *reinterpret_cast<const v16bf*>(&wr_sw[(ch * 32 + lane) * 16]);
        accs[ch & 3] = wmma_bf16(a, b, accs[ch & 3]);
      }
      const v8f acc = (accs[0] + accs[1]) + (accs[2] + accs[3]);

      if (half == 0) { // lanes 0-15 hold batches 0..7 in VGPRs 0..7
#pragma unroll
        for (int r = 0; r < 8; ++r) {
          const size_t row = (size_t)r * Ll + t;
          const float pred = acc[r] + brn;
          const float hnew = df * hold[r] + omd * (zv[r] * (xv[r] - pred));
          hnxt[r * Dd + ncol]  = hnew;
          ybuf[row * Dd + ncol] = hnew;
        }
      }
    } else if (tid >= 32 && tid < 64 && t + 8 < Ll) {
      // wave 1: prefetch x and z rows for step t+8 (HBM -> L2).
      const int pl = tid & 31;
      const int pc = n0 + (pl & 15);
      const int pg = pl >> 4; // 2 groups of 4 batches
#pragma unroll
      for (int rr = 0; rr < 4; ++rr) {
        const size_t row = (size_t)(pg * 4 + rr) * Ll + (t + 8);
        __builtin_prefetch(&x[row * Dd + pc], 0, 1);
        __builtin_prefetch(&zbuf[row * Dd + pc], 0, 1);
      }
    }

    // Device-wide barrier (monotonic counter; reset each launch).
    __threadfence();
    __syncthreads();
    if (tid == 0) {
      __hip_atomic_fetch_add(cnt, 1u, __ATOMIC_RELEASE, __HIP_MEMORY_SCOPE_AGENT);
      const unsigned target = (unsigned)(t + 1) * nblk;
      while (__hip_atomic_load(cnt, __ATOMIC_ACQUIRE, __HIP_MEMORY_SCOPE_AGENT) < target)
        __builtin_amdgcn_s_sleep(2);
      __threadfence();
    }
    __syncthreads();
  }
}

// ---------------------------------------------------------------------------
// Row LayerNorm: y_ln = (y-mean)*rsqrt(var+eps)*ln_w + ln_b -> bf16, written
// in fragment-swizzled order so the output GEMM can async-copy it directly.
// One 256-thread block per row; each thread owns 4 consecutive features.
// ---------------------------------------------------------------------------
__global__ void ln_kernel(const float* __restrict__ y,
                          const float* __restrict__ ln_w,
                          const float* __restrict__ ln_b,
                          __bf16* __restrict__ yl_swz) {
  __shared__ float rs[256];
  __shared__ float rs2[256];
  const int tid = threadIdx.x;
  const size_t row = blockIdx.x;
  const float* yr = y + row * Dd;
  const int k = tid * 4;

  const float4 v = *reinterpret_cast<const float4*>(yr + k);
  rs[tid]  = v.x + v.y + v.z + v.w;
  rs2[tid] = v.x * v.x + v.y * v.y + v.z * v.z + v.w * v.w;
  __syncthreads();
  for (int off = 128; off > 0; off >>= 1) {
    if (tid < off) { rs[tid] += rs[tid + off]; rs2[tid] += rs2[tid + off]; }
    __syncthreads();
  }
  const float mean = rs[0] * (1.0f / Dd);
  const float var  = rs2[0] * (1.0f / Dd) - mean * mean;
  const float inv  = rsqrtf(var + 1e-5f);

  __bf16* d = yl_swz + aswz((int)row, k); // 4 consecutive slots
  const float4 w = *reinterpret_cast<const float4*>(ln_w + k);
  const float4 b = *reinterpret_cast<const float4*>(ln_b + k);
  d[0] = (__bf16)((v.x - mean) * inv * w.x + b.x);
  d[1] = (__bf16)((v.y - mean) * inv * w.y + b.y);
  d[2] = (__bf16)((v.z - mean) * inv * w.z + b.z);
  d[3] = (__bf16)((v.w - mean) * inv * w.w + b.w);
}

// ---------------------------------------------------------------------------
// Host launch. ws layout:
//   [0, 2M)      Wg bf16 swizzled   [2M, 4M) Wr swz   [4M, 6M) Wo swz
//   [6M, 70M)    z bf16 row-major (32768x1024), reused as swizzled y_ln
//   [70M, +64K)  h ping-pong f32 (2 x 8 x 1024), then barrier counter
// d_out doubles as f32 y staging between scan and final GEMM.
// ---------------------------------------------------------------------------
extern "C" void kernel_launch(void* const* d_in, const int* in_sizes, int n_in,
                              void* d_out, int out_size, void* d_ws, size_t ws_size,
                              hipStream_t stream) {
  const float* x     = (const float*)d_in[0];
  const float* decay = (const float*)d_in[1];
  const float* Wr    = (const float*)d_in[2];
  const float* br    = (const float*)d_in[3];
  const float* Wg    = (const float*)d_in[4];
  const float* bg    = (const float*)d_in[5];
  const float* Wo    = (const float*)d_in[6];
  const float* bo    = (const float*)d_in[7];
  const float* ln_w  = (const float*)d_in[8];
  const float* ln_b  = (const float*)d_in[9];
  float* out = (float*)d_out;

  const size_t WSZ = (size_t)Dd * Dd;
  __bf16* Wg_swz = (__bf16*)d_ws;
  __bf16* Wr_swz = Wg_swz + WSZ;
  __bf16* Wo_swz = Wr_swz + WSZ;
  __bf16* zbuf   = Wo_swz + WSZ;                     // 64 MB, reused as y_ln
  float*  hbuf   = (float*)(zbuf + (size_t)Mrows * Dd);
  unsigned* cnt  = (unsigned*)(hbuf + 2 * Bb * Dd);

  to_bf16_swz_kernel<<<512, 256, 0, stream>>>(Wg, Wg_swz);
  to_bf16_swz_kernel<<<512, 256, 0, stream>>>(Wr, Wr_swz);
  to_bf16_swz_kernel<<<512, 256, 0, stream>>>(Wo, Wo_swz);
  reset_kernel<<<64, 256, 0, stream>>>(hbuf, cnt);

  dim3 ggrid(Mrows / 32, Dd / 64);
  // z = sigmoid(x @ Wg^T + bg)
  gemm_kernel<0><<<ggrid, 256, 0, stream>>>(x, nullptr, Wg_swz, bg, zbuf, nullptr);
  // recurrent scan: y (f32) staged into d_out
  scan_kernel<<<64, 128, 0, stream>>>(x, decay, br, Wr_swz, zbuf, hbuf, cnt, out);
  // layernorm y -> swizzled y_ln (reuses z buffer)
  ln_kernel<<<Mrows, 256, 0, stream>>>(out, ln_w, ln_b, zbuf);
  // out = y_ln @ Wo^T + bo
  gemm_kernel<1><<<ggrid, 256, 0, stream>>>(nullptr, zbuf, Wo_swz, bo, nullptr, out);
}
